// MultiHeadAttention_19559281066284
// MI455X (gfx1250) — matrix operand order repaired and checked by reading, not compiled, not run on hardware
//
#include <hip/hip_runtime.h>
#include <hip/hip_bf16.h>
#include <stdint.h>

#define D_MODEL 1024
#define NUM_HEADS 16
#define D_K 64
#define SEQ 2048
#define BATCH 4
#define M_TOT (BATCH * SEQ)   // 8192 rows

typedef __bf16 bf16;
typedef __bf16 v16bf __attribute__((ext_vector_type(16)));
typedef __bf16 v8bf  __attribute__((ext_vector_type(8)));
typedef float  v8f   __attribute__((ext_vector_type(8)));
typedef unsigned int u32x4 __attribute__((ext_vector_type(4)));
typedef int i32x4 __attribute__((ext_vector_type(4)));
typedef int i32x8 __attribute__((ext_vector_type(8)));

#if defined(__HIP_DEVICE_COMPILE__) && __has_builtin(__builtin_amdgcn_tensor_load_to_lds)
#define HAVE_TDM 1
#else
#define HAVE_TDM 0
#endif

static __device__ __forceinline__ v16bf cat8(v8bf lo, v8bf hi) {
    return __builtin_shufflevector(lo, hi, 0,1,2,3,4,5,6,7,8,9,10,11,12,13,14,15);
}
// 32B fragment from a 16B-aligned LDS address (two b128 reads)
static __device__ __forceinline__ v16bf ld16(const void* p) {
    return cat8(*(const v8bf*)p, *(const v8bf*)((const unsigned char*)p + 16));
}
static __device__ __forceinline__ v8f wmma_bf16(v16bf a, v16bf b, v8f c) {
    return __builtin_amdgcn_wmma_f32_16x16x32_bf16(false, a, false, b, (short)0, c, false, false);
}
static __device__ __forceinline__ uint32_t lds_byte(const void* p) {
    // LDS aperture: flat address low 32 bits == LDS byte address
    return (uint32_t)(uintptr_t)p;
}
static __device__ __forceinline__ void wait_tensorcnt0() {
#if HAVE_TDM
#if __has_builtin(__builtin_amdgcn_s_wait_tensorcnt)
    __builtin_amdgcn_s_wait_tensorcnt(0);
#else
    asm volatile("s_wait_tensorcnt 0x0" ::: "memory");
#endif
#endif
}

#if HAVE_TDM
// 2D tile load via Tensor Data Mover. Dims/strides in elements (2B each).
// padC: pad after 8<<c bytes; padA: pad of 4*(a+1) bytes.
static __device__ __forceinline__ void tdm_load_2d(const void* gsrc, uint32_t ldsByte,
        uint32_t tdim0, uint32_t tdim1, uint32_t tile0, uint32_t tile1,
        uint64_t stride0, uint32_t padC, uint32_t padA) {
    const uint64_t ga = (uint64_t)(uintptr_t)gsrc;
    u32x4 g0;
    g0.x = 1u;                                                 // count=1, user mode
    g0.y = ldsByte;                                            // lds_addr
    g0.z = (uint32_t)ga;                                       // global_addr lo
    g0.w = ((uint32_t)(ga >> 32) & 0x01FFFFFFu) | 0x80000000u; // addr hi | type=2
    i32x8 g1;
    g1[0] = (int)((1u << 16) | (1u << 20) | (padC << 22) | (padA << 25)); // 2B elems, pad_en
    g1[1] = (int)((tdim0 & 0xFFFFu) << 16);
    g1[2] = (int)((tdim0 >> 16) | ((tdim1 & 0xFFFFu) << 16));
    g1[3] = (int)((tdim1 >> 16) | ((tile0 & 0xFFFFu) << 16));
    g1[4] = (int)(tile1 & 0xFFFFu);                            // tile_dim2 = 0
    g1[5] = (int)(uint32_t)stride0;
    g1[6] = (int)((uint32_t)(stride0 >> 32) & 0xFFFFu);        // stride1 = 0
    g1[7] = 0;
    i32x4 z4 = {0, 0, 0, 0};
#if defined(__clang_major__) && __clang_major__ >= 23
    i32x8 z8 = {0, 0, 0, 0, 0, 0, 0, 0};
    __builtin_amdgcn_tensor_load_to_lds(g0, g1, z4, z4, z8, 0);
#else
    __builtin_amdgcn_tensor_load_to_lds(g0, g1, z4, z4, 0);
#endif
}
#endif

// ---------------------------------------------------------------- fp32 -> bf16
__global__ void cvt_f32_bf16(const float* __restrict__ in, bf16* __restrict__ out, int n) {
    int i = blockIdx.x * blockDim.x + threadIdx.x;
    if (i < n) out[i] = (bf16)in[i];
}

// ---------------------------------------------------------------- WMMA GEMM
// Y[m,n] = sum_k A[m,k] * W[n,k] (+ bias). One block = one 16-row M tile (staged
// once via TDM into LDS); each of the 8 waves computes a 16x128 slab of N
// (8 WMMAs per k-step against one shared A fragment -> ~16 flop/L2-byte).
#define A_ROWB 2080u   // 2048B row + 2x16B TDM pads (every 1024B)
static __device__ __forceinline__ uint32_t a_off(int row, int col) {
    return (uint32_t)row * A_ROWB + (uint32_t)col * 2u + ((col >= 512) ? 16u : 0u);
}

__global__ void gemm_bf16(const bf16* __restrict__ A, const bf16* __restrict__ W,
                          const float* __restrict__ bias,
                          bf16* __restrict__ outb, float* __restrict__ outf,
                          const float* __restrict__ resid, int mode) {
    __shared__ __align__(32) unsigned char atile[16 * A_ROWB];   // 33.3 KB
    const int tid    = threadIdx.x;
    const int lane   = tid & 31;
    const int wid    = tid >> 5;
    const int m_base = blockIdx.x * 16;
    const int n_base = wid * 128;
    const int ln     = lane & 15;
    const int half   = lane >> 4;

#if HAVE_TDM
    if (tid < 32) {   // wave 0 issues one TDM for the whole 16x1024 A tile
        tdm_load_2d(A + (size_t)m_base * D_MODEL, lds_byte(atile),
                    D_MODEL, 16, D_MODEL, 16, D_MODEL, /*pad every 1024B*/7, /*16B*/3);
        wait_tensorcnt0();
    }
#else
    for (int i = tid; i < 16 * (D_MODEL / 8); i += 256) {
        int r = i >> 7, c = (i & 127) * 8;
        *(v8bf*)(atile + a_off(r, c)) = *(const v8bf*)(A + (size_t)(m_base + r) * D_MODEL + c);
    }
#endif
    __syncthreads();

    v8f acc[8];
#pragma unroll
    for (int j = 0; j < 8; ++j) acc[j] = (v8f){};
    const bf16* wr[8];
#pragma unroll
    for (int j = 0; j < 8; ++j)
        wr[j] = W + (size_t)(n_base + j * 16 + ln) * D_MODEL + half * 16;

    for (int k = 0; k < D_MODEL; k += 32) {
        __builtin_prefetch(wr[0] + k + 256, 0, 3);
        v16bf af = cat8(*(const v8bf*)(atile + a_off(ln, k + half * 16)),
                        *(const v8bf*)(atile + a_off(ln, k + half * 16 + 8)));
#pragma unroll
        for (int j = 0; j < 8; ++j) {
            v16bf bj = *(const v16bf*)(wr[j] + k);
            acc[j] = wmma_bf16(af, bj, acc[j]);
        }
    }

#pragma unroll
    for (int j = 0; j < 8; ++j) {
        const int ng = n_base + j * 16 + ln;
        const float bv = bias[ng];
#pragma unroll
        for (int r = 0; r < 8; ++r) {
            const int mg = m_base + r + half * 8;
            float v = acc[j][r] + bv;
            if (mode == 3) {
                outf[(size_t)mg * D_MODEL + ng] = v + resid[(size_t)mg * D_MODEL + ng];
            } else {
                const int bb = mg >> 11, s = mg & (SEQ - 1);
                const int h = ng >> 6, dk = ng & 63;
                if (mode == 2)
                    outb[(size_t)((bb * NUM_HEADS + h) * D_K + dk) * SEQ + s] = (bf16)v;
                else
                    outb[(size_t)((bb * NUM_HEADS + h) * SEQ + s) * D_K + dk] = (bf16)v;
            }
        }
    }
}

// ---------------------------------------------------------------- flash attention
// 8 waves/block share one (b,h): K/V 32-key tiles staged via TDM, double-buffered.
#define K_ROWB 144u                 // 128B row + 16B pad (bank-conflict-free)
#define V_ROWB 80u                  // 64B row + 16B pad
#define KSLAB  (32u * K_ROWB)       // 4608 B
#define VSLAB  (64u * V_ROWB)       // 5120 B
#define P_ROW  40                   // P-buffer row stride in elems (80B)

static __device__ __forceinline__ void flash_stage(unsigned char* kslab, unsigned char* vslab,
        const bf16* Kh, const bf16* Vh, int kb, int tid) {
#if HAVE_TDM
    if (tid < 32) {
        // K tile: 32 rows x 64 elems, row stride D_K; pad every 128B by 16B
        tdm_load_2d(Kh + (size_t)kb * D_K, lds_byte(kslab), D_K, 32, D_K, 32, D_K, 4, 3);
        // V tile: 64 rows x 32 elems from Vt, row stride SEQ; pad every 64B by 16B
        tdm_load_2d(Vh + kb, lds_byte(vslab), 32, D_K, 32, D_K, SEQ, 3, 3);
    }
#else
    {
        int r = tid >> 3, c = tid & 7;          // 32 rows x 8 chunks of 16B
        *(v8bf*)(kslab + r * K_ROWB + c * 16) =
            *(const v8bf*)(Kh + (size_t)(kb + r) * D_K + c * 8);
        int vr = tid >> 2, vc = tid & 3;        // 64 rows x 4 chunks of 16B
        *(v8bf*)(vslab + vr * V_ROWB + vc * 16) =
            *(const v8bf*)(Vh + (size_t)vr * SEQ + kb + vc * 8);
    }
#endif
}

__global__ void flash_attn(const bf16* __restrict__ Qb, const bf16* __restrict__ Kb,
                           const bf16* __restrict__ Vt, bf16* __restrict__ ctx) {
    __shared__ __align__(32) unsigned char kbuf[2 * KSLAB];
    __shared__ __align__(32) unsigned char vbuf[2 * VSLAB];
    __shared__ __align__(32) bf16 pbuf[8 * 16 * P_ROW];
    const int tid   = threadIdx.x;
    const int lane  = tid & 31;
    const int wid   = tid >> 5;
    const int gw    = blockIdx.x * 8 + wid;
    const int qtile = gw & 127;
    const int h     = (gw >> 7) & (NUM_HEADS - 1);
    const int b     = gw >> 11;
    const int ln    = lane & 15;
    const int half  = lane >> 4;

    const bf16* Qh = Qb + (size_t)(b * NUM_HEADS + h) * SEQ * D_K;
    const bf16* Kh = Kb + (size_t)(b * NUM_HEADS + h) * SEQ * D_K;
    const bf16* Vh = Vt + (size_t)(b * NUM_HEADS + h) * D_K * SEQ;
    bf16* pb = pbuf + wid * (16 * P_ROW);

    flash_stage(kbuf, vbuf, Kh, Vh, 0, tid);    // prologue: slab 0

    const bf16* qrow = Qh + (size_t)(qtile * 16 + ln) * D_K;
    v16bf qf0 = cat8(*(const v8bf*)(qrow + half * 16),
                     *(const v8bf*)(qrow + half * 16 + 8));
    v16bf qf1 = cat8(*(const v8bf*)(qrow + 32 + half * 16),
                     *(const v8bf*)(qrow + 32 + half * 16 + 8));

    v8f c0 = {}, c1 = {}, c2 = {}, c3 = {};
    float mrun[8], lrun[8];
#pragma unroll
    for (int r = 0; r < 8; ++r) { mrun[r] = -3.0e38f; lrun[r] = 0.f; }
    const float SCALE = 0.125f;   // 1/sqrt(64)

    for (int kb = 0; kb < SEQ; kb += 32) {
        const int cur = (kb >> 5) & 1;
#if HAVE_TDM
        if (tid < 32) wait_tensorcnt0();        // wave0's TDM for slab `cur` done
#endif
        __syncthreads();                        // tiles visible; prev slab free
        if (kb + 32 < SEQ)
            flash_stage(kbuf + (cur ^ 1) * KSLAB, vbuf + (cur ^ 1) * VSLAB,
                        Kh, Vh, kb + 32, tid);

        const unsigned char* kc = kbuf + cur * KSLAB;
        const unsigned char* vc = vbuf + cur * VSLAB;

        // S = Q @ K^T (two 16x16 tiles, dk = 64 -> two K=32 wmma steps each)
        v16bf bk00 = ld16(kc + (     ln) * K_ROWB + half * 32);
        v16bf bk01 = ld16(kc + (     ln) * K_ROWB + 64 + half * 32);
        v16bf bk10 = ld16(kc + (16 + ln) * K_ROWB + half * 32);
        v16bf bk11 = ld16(kc + (16 + ln) * K_ROWB + 64 + half * 32);
        v8f s0 = {}; s0 = wmma_bf16(qf0, bk00, s0); s0 = wmma_bf16(qf1, bk01, s0);
        v8f s1 = {}; s1 = wmma_bf16(qf0, bk10, s1); s1 = wmma_bf16(qf1, bk11, s1);

#pragma unroll
        for (int r = 0; r < 8; ++r) {
            float e0 = s0[r] * SCALE, e1 = s1[r] * SCALE;
            float mx = fmaxf(e0, e1);
            mx = fmaxf(mx, __shfl_xor(mx, 1));
            mx = fmaxf(mx, __shfl_xor(mx, 2));
            mx = fmaxf(mx, __shfl_xor(mx, 4));
            mx = fmaxf(mx, __shfl_xor(mx, 8));
            const float mnew = fmaxf(mrun[r], mx);
            const float corr = __expf(mrun[r] - mnew);
            mrun[r] = mnew;
            e0 = __expf(e0 - mnew); e1 = __expf(e1 - mnew);
            float ps = e0 + e1;
            ps += __shfl_xor(ps, 1); ps += __shfl_xor(ps, 2);
            ps += __shfl_xor(ps, 4); ps += __shfl_xor(ps, 8);
            lrun[r] = lrun[r] * corr + ps;
            c0[r] *= corr; c1[r] *= corr; c2[r] *= corr; c3[r] *= corr;
            const int mrow = r + half * 8;                  // C-layout row
            pb[mrow * P_ROW + ln]      = (bf16)e0;
            pb[mrow * P_ROW + 16 + ln] = (bf16)e1;
        }
        asm volatile("s_wait_dscnt 0" ::: "memory");
        // Re-read P as a 16x32 A-fragment (K = key index)
        v16bf pf = cat8(*(const v8bf*)(pb + ln * P_ROW + half * 16),
                        *(const v8bf*)(pb + ln * P_ROW + half * 16 + 8));
        // ctx += P @ V
        v16bf bv0 = ld16(vc + ( 0 + ln) * V_ROWB + half * 32);
        v16bf bv1 = ld16(vc + (16 + ln) * V_ROWB + half * 32);
        v16bf bv2 = ld16(vc + (32 + ln) * V_ROWB + half * 32);
        v16bf bv3 = ld16(vc + (48 + ln) * V_ROWB + half * 32);
        c0 = wmma_bf16(pf, bv0, c0);
        c1 = wmma_bf16(pf, bv1, c1);
        c2 = wmma_bf16(pf, bv2, c2);
        c3 = wmma_bf16(pf, bv3, c3);
    }

#pragma unroll
    for (int r = 0; r < 8; ++r) {
        const float inv = 1.0f / lrun[r];
        const size_t mg = (size_t)(b * SEQ + qtile * 16 + r + half * 8);
        ctx[mg * D_MODEL + h * D_K +  0 + ln] = (bf16)(c0[r] * inv);
        ctx[mg * D_MODEL + h * D_K + 16 + ln] = (bf16)(c1[r] * inv);
        ctx[mg * D_MODEL + h * D_K + 32 + ln] = (bf16)(c2[r] * inv);
        ctx[mg * D_MODEL + h * D_K + 48 + ln] = (bf16)(c3[r] * inv);
    }
}

// ---------------------------------------------------------------- LayerNorm
__global__ void ln_kernel(const float* __restrict__ y, const float* __restrict__ gamma,
                          const float* __restrict__ beta, float* __restrict__ out) {
    __shared__ float rs[256], rs2[256];
    const int row = blockIdx.x, tid = threadIdx.x;
    const float* yr = y + (size_t)row * D_MODEL;
    float s = 0.f, s2 = 0.f;
    for (int i = tid; i < D_MODEL; i += 256) { float v = yr[i]; s += v; s2 += v * v; }
    rs[tid] = s; rs2[tid] = s2; __syncthreads();
    for (int st = 128; st > 0; st >>= 1) {
        if (tid < st) { rs[tid] += rs[tid + st]; rs2[tid] += rs2[tid + st]; }
        __syncthreads();
    }
    const float mu = rs[0] * (1.0f / D_MODEL);
    const float var = rs2[0] * (1.0f / D_MODEL) - mu * mu;
    const float rstd = rsqrtf(var + 1e-5f);
    for (int i = tid; i < D_MODEL; i += 256)
        out[(size_t)row * D_MODEL + i] = (yr[i] - mu) * rstd * gamma[i] + beta[i];
}

// ---------------------------------------------------------------- launch
extern "C" void kernel_launch(void* const* d_in, const int* in_sizes, int n_in,
                              void* d_out, int out_size, void* d_ws, size_t ws_size,
                              hipStream_t stream) {
    const float* x     = (const float*)d_in[0];
    const float* Wq    = (const float*)d_in[1];
    const float* bq    = (const float*)d_in[2];
    const float* Wk    = (const float*)d_in[3];
    const float* bk    = (const float*)d_in[4];
    const float* Wv    = (const float*)d_in[5];
    const float* bv    = (const float*)d_in[6];
    const float* Wo    = (const float*)d_in[7];
    const float* bo    = (const float*)d_in[8];
    const float* gamma = (const float*)d_in[9];
    const float* beta  = (const float*)d_in[10];
    float* out = (float*)d_out;

    char* ws = (char*)d_ws;
    size_t off = 0;
    const size_t NX = (size_t)M_TOT * D_MODEL;
    const size_t NW = (size_t)D_MODEL * D_MODEL;
    bf16* xb   = (bf16*)(ws + off); off += NX * 2;
    bf16* wqb  = (bf16*)(ws + off); off += NW * 2;
    bf16* wkb  = (bf16*)(ws + off); off += NW * 2;
    bf16* wvb  = (bf16*)(ws + off); off += NW * 2;
    bf16* wob  = (bf16*)(ws + off); off += NW * 2;
    bf16* Qbf  = (bf16*)(ws + off); off += NX * 2;   // [b,h,s,dk]
    bf16* Kbf  = (bf16*)(ws + off); off += NX * 2;   // [b,h,s,dk]
    bf16* Vtb  = (bf16*)(ws + off); off += NX * 2;   // [b,h,dk,s]
    bf16* ctxb = (bf16*)(ws + off); off += NX * 2;   // [m, e]
    float* y   = (float*)(ws + off); off += NX * 4;  // pre-LN

    cvt_f32_bf16<<<(int)(NX / 256), 256, 0, stream>>>(x,  xb,  (int)NX);
    cvt_f32_bf16<<<(int)(NW / 256), 256, 0, stream>>>(Wq, wqb, (int)NW);
    cvt_f32_bf16<<<(int)(NW / 256), 256, 0, stream>>>(Wk, wkb, (int)NW);
    cvt_f32_bf16<<<(int)(NW / 256), 256, 0, stream>>>(Wv, wvb, (int)NW);
    cvt_f32_bf16<<<(int)(NW / 256), 256, 0, stream>>>(Wo, wob, (int)NW);

    // one block per 16-row M tile; each covers the full N=1024
    gemm_bf16<<<M_TOT / 16, 256, 0, stream>>>(xb, wqb, bq, Qbf, nullptr, nullptr, 0);
    gemm_bf16<<<M_TOT / 16, 256, 0, stream>>>(xb, wkb, bk, Kbf, nullptr, nullptr, 0);
    gemm_bf16<<<M_TOT / 16, 256, 0, stream>>>(xb, wvb, bv, Vtb, nullptr, nullptr, 2);

    flash_attn<<<1024, 256, 0, stream>>>(Qbf, Kbf, Vtb, ctxb);

    gemm_bf16<<<M_TOT / 16, 256, 0, stream>>>(ctxb, wob, bo, nullptr, y, x, 3);

    ln_kernel<<<M_TOT, 256, 0, stream>>>(y, gamma, beta, out);
}
